// MambaBlockCPU_86595130622294
// MI455X (gfx1250) — matrix operand order repaired and checked by reading, compiled, not run on hardware
//
#include <hip/hip_runtime.h>

// ---------------------------------------------------------------------------
// Mamba block forward for MI455X (gfx1250).
// bf16 WMMA GEMMs (async global->LDS staging for A tiles) + fused f32 scan.
// ---------------------------------------------------------------------------

#define D_MODEL   1024
#define D_STATE   16
#define D_CONV    4
#define D_INNER   2048
#define DT_RANK   64
#define BATCH     2
#define SEQL      2048
#define NTOK      (BATCH * SEQL)          // 4096 rows for all GEMMs

typedef __bf16           v16bf  __attribute__((ext_vector_type(16)));
typedef float            v8f    __attribute__((ext_vector_type(8)));
typedef unsigned short   v8u16  __attribute__((ext_vector_type(8)));
typedef unsigned short   v16u16 __attribute__((ext_vector_type(16)));

__device__ __forceinline__ unsigned short f32_to_bf16(float f) {
  unsigned int u = __float_as_uint(f);
  u += 0x7FFFu + ((u >> 16) & 1u);          // round-to-nearest-even
  return (unsigned short)(u >> 16);
}

__device__ __forceinline__ float silu_f(float v) {
  return v / (1.0f + __expf(-v));
}

// CDNA5 async copy: global memory -> LDS, 16 bytes per lane, ASYNCcnt-tracked.
__device__ __forceinline__ void async_load_b128(unsigned lds_addr, const void* gaddr) {
  asm volatile("global_load_async_to_lds_b128 %0, %1, off"
               :
               : "v"(lds_addr), "v"((unsigned long long)gaddr)
               : "memory");
}

__device__ __forceinline__ void wait_async0() {
  asm volatile("s_wait_asynccnt 0x0" ::: "memory");
}

// ---------------------------------------------------------------------------
// Generic GEMM: C(MxN,f32) = A(MxK,bf16 row-major) * B(KxN,bf16 row-major)
// Requirements: M % 128 == 0, K % 32 == 0, N % 8 == 0 (N edges guarded).
// Block = 256 threads = 8 waves; block tile 128x64; wave tile 32x32 (4 WMMAs).
// A tile staged with GLOBAL_LOAD_ASYNC_TO_LDS_B128; B tile transposed in LDS.
// ---------------------------------------------------------------------------
__global__ void __launch_bounds__(256)
gemm_bf16_wmma(const unsigned short* __restrict__ A,
               const unsigned short* __restrict__ Bm,
               float* __restrict__ C,
               int M, int N, int K)
{
  __shared__ __align__(16) unsigned short As[128][40];  // [m][k], +8 pad
  __shared__ __align__(16) unsigned short Bt[64][40];   // [n][k] transposed

  const int tid  = threadIdx.x;
  const int m0   = blockIdx.x * 128;
  const int n0   = blockIdx.y * 64;
  const int wave = tid >> 5;
  const int lane = tid & 31;
  const int wm   = wave & 3;          // 0..3 -> 32-row slab
  const int wn   = wave >> 2;         // 0..1 -> 32-col slab

  // cooperative tile-load mapping
  const int ar = tid >> 1;            // A row within tile (0..127)
  const int ac = (tid & 1) << 4;      // A col chunk (0 or 16)
  const int kb = tid >> 3;            // B k-row within tile (0..31)
  const int nb = (tid & 7) << 3;      // B n chunk (0..56)

  const unsigned ldsA0 = (unsigned)(size_t)&As[ar][ac];
  const unsigned ldsA1 = ldsA0 + 16;

  // WMMA fragment mapping (ISA 16-bit A 16x32 / B 32x16 layouts)
  const int mrow0 = wm * 32 + (lane & 15);
  const int koff  = (lane >> 4) << 3;   // 0 or 8
  const int ncol0 = wn * 32 + (lane & 15);
  const int kbase = (lane >> 4) << 3;   // 0 or 8

  v8f acc00 = {}; v8f acc01 = {}; v8f acc10 = {}; v8f acc11 = {};

  for (int k0 = 0; k0 < K; k0 += 32) {
    // A tile: async global->LDS (2 x 16B per thread), no VGPR round-trip
    const unsigned short* ga = A + (size_t)(m0 + ar) * K + (k0 + ac);
    async_load_b128(ldsA0, ga);
    async_load_b128(ldsA1, ga + 8);

    // B tile: registers + transpose scatter into Bt[n][k]
    v8u16 bv = {};
    const bool bok = (n0 + nb) < N;
    if (bok) bv = *(const v8u16*)(Bm + (size_t)(k0 + kb) * N + (n0 + nb));
#pragma unroll
    for (int j = 0; j < 8; ++j) Bt[nb + j][kb] = bv[j];

    if (k0 + 32 < K) {   // prefetch next K-tile -> global_prefetch_b8
      __builtin_prefetch(A + (size_t)(m0 + ar) * K + (k0 + 32 + ac), 0, 1);
      if (bok)
        __builtin_prefetch(Bm + (size_t)(k0 + 32 + kb) * N + (n0 + nb), 0, 1);
    }

    wait_async0();           // async A copies landed in LDS
    __syncthreads();

    // A fragments: K chunks {koff..koff+7} and {16+koff..16+koff+7}
    v8u16 a0lo = *(const v8u16*)&As[mrow0][koff];
    v8u16 a0hi = *(const v8u16*)&As[mrow0][16 + koff];
    v8u16 a1lo = *(const v8u16*)&As[mrow0 + 16][koff];
    v8u16 a1hi = *(const v8u16*)&As[mrow0 + 16][16 + koff];
    v16u16 au0, au1;
#pragma unroll
    for (int j = 0; j < 8; ++j) {
      au0[j] = a0lo[j]; au0[8 + j] = a0hi[j];
      au1[j] = a1lo[j]; au1[8 + j] = a1hi[j];
    }
    v16bf af0 = __builtin_bit_cast(v16bf, au0);
    v16bf af1 = __builtin_bit_cast(v16bf, au1);

    // B fragments: K chunks {kbase..kbase+7} and {16+kbase..16+kbase+7} (transposed LDS)
    v8u16 b0lo = *(const v8u16*)&Bt[ncol0][kbase];
    v8u16 b0hi = *(const v8u16*)&Bt[ncol0][kbase + 16];
    v8u16 b1lo = *(const v8u16*)&Bt[ncol0 + 16][kbase];
    v8u16 b1hi = *(const v8u16*)&Bt[ncol0 + 16][kbase + 16];
    v16u16 bu0, bu1;
#pragma unroll
    for (int j = 0; j < 8; ++j) {
      bu0[j] = b0lo[j]; bu0[8 + j] = b0hi[j];
      bu1[j] = b1lo[j]; bu1[8 + j] = b1hi[j];
    }
    v16bf bf0 = __builtin_bit_cast(v16bf, bu0);
    v16bf bf1 = __builtin_bit_cast(v16bf, bu1);

    acc00 = __builtin_amdgcn_wmma_f32_16x16x32_bf16(false, af0, false, bf0,
                                                    (short)0, acc00, false, false);
    acc01 = __builtin_amdgcn_wmma_f32_16x16x32_bf16(false, af0, false, bf1,
                                                    (short)0, acc01, false, false);
    acc10 = __builtin_amdgcn_wmma_f32_16x16x32_bf16(false, af1, false, bf0,
                                                    (short)0, acc10, false, false);
    acc11 = __builtin_amdgcn_wmma_f32_16x16x32_bf16(false, af1, false, bf1,
                                                    (short)0, acc11, false, false);
    __syncthreads();
  }

  // C/D layout: VGPR r -> M = r (lanes 0-15) or 8+r (lanes 16-31); N = lane&15
  const int crow = m0 + wm * 32 + ((lane >> 4) << 3);
  const int cc0  = n0 + wn * 32 + (lane & 15);
#pragma unroll
  for (int r = 0; r < 8; ++r) {
    if (cc0 < N) {
      C[(size_t)(crow + r) * N + cc0]      = acc00[r];
      C[(size_t)(crow + 16 + r) * N + cc0] = acc10[r];
    }
    if (cc0 + 16 < N) {
      C[(size_t)(crow + r) * N + cc0 + 16]      = acc01[r];
      C[(size_t)(crow + 16 + r) * N + cc0 + 16] = acc11[r];
    }
  }
}

// ---------------------------------------------------------------------------
// f32 -> bf16 conversion
// ---------------------------------------------------------------------------
__global__ void cvt_f32_bf16(const float* __restrict__ in,
                             unsigned short* __restrict__ out, int n)
{
  int i = blockIdx.x * 256 + threadIdx.x;
  if (i < n) out[i] = f32_to_bf16(in[i]);
}

// ---------------------------------------------------------------------------
// Causal depthwise conv1d (width 4, left pad 3) + bias + SiLU.
// xz layout: (B*L, 4096), x_inner = cols [0,2048).
// Writes x_conv in f32 (for scan) and bf16 (for x_dbl GEMM).
// ---------------------------------------------------------------------------
__global__ void conv_silu(const float* __restrict__ xz,
                          const float* __restrict__ conv_w,   // (2048,1,4)
                          const float* __restrict__ conv_b,   // (2048)
                          float* __restrict__ xc_f32,
                          unsigned short* __restrict__ xc_bf16)
{
  int idx = blockIdx.x * 256 + threadIdx.x;
  if (idx >= BATCH * SEQL * D_INNER) return;
  int d = idx & (D_INNER - 1);
  int t = (idx / D_INNER) & (SEQL - 1);
  int b = idx / (D_INNER * SEQL);
  const float* w = conv_w + d * D_CONV;
  float acc = conv_b[d];
#pragma unroll
  for (int k = 0; k < D_CONV; ++k) {
    int tt = t - (D_CONV - 1) + k;
    if (tt >= 0)
      acc += w[k] * xz[((size_t)(b * SEQL + tt)) * (2 * D_INNER) + d];
  }
  float s = silu_f(acc);
  xc_f32[idx]  = s;
  xc_bf16[idx] = f32_to_bf16(s);
}

// ---------------------------------------------------------------------------
// Extract dt columns (0..63) of x_dbl (rows x 96) as bf16 GEMM-A operand.
// ---------------------------------------------------------------------------
__global__ void extract_dt_bf16(const float* __restrict__ xdbl,
                                unsigned short* __restrict__ out, int rows)
{
  int i = blockIdx.x * 256 + threadIdx.x;
  if (i >= rows * DT_RANK) return;
  int r = i >> 6, c = i & 63;
  out[i] = f32_to_bf16(xdbl[r * (DT_RANK + 2 * D_STATE) + c]);
}

// ---------------------------------------------------------------------------
// Fused selective scan: softplus(dt+b_dt), h recurrence (f32 regs),
// y + x*D, gate with silu(z), emit bf16 for the output GEMM.
// Grid: (D_INNER/256, BATCH); each thread owns one channel's h[16].
// B_t/C_t staged in LDS in 32-step chunks.
// ---------------------------------------------------------------------------
#define TCHUNK 32
__global__ void __launch_bounds__(256)
scan_fused(const float* __restrict__ xdbl,     // (B*L, 96): [64:80]=B, [80:96]=C
           const float* __restrict__ dt_lin,   // (B*L, 2048)
           const float* __restrict__ b_dt,     // (2048)
           const float* __restrict__ xconv,    // (B*L, 2048)
           const float* __restrict__ xz,       // (B*L, 4096): z = cols [2048,4096)
           const float* __restrict__ A_log,    // (2048,16)
           const float* __restrict__ Dvec,     // (2048)
           unsigned short* __restrict__ ygate) // (B*L, 2048) bf16
{
  __shared__ float Bs[TCHUNK][D_STATE];
  __shared__ float Cs[TCHUNK][D_STATE];

  const int d = blockIdx.x * 256 + threadIdx.x;   // channel
  const int b = blockIdx.y;

  float Arow[D_STATE];
#pragma unroll
  for (int n = 0; n < D_STATE; ++n) Arow[n] = -__expf(A_log[d * D_STATE + n]);
  const float Dv  = Dvec[d];
  const float bdt = b_dt[d];

  float h[D_STATE];
#pragma unroll
  for (int n = 0; n < D_STATE; ++n) h[n] = 0.0f;

  for (int t0 = 0; t0 < SEQL; t0 += TCHUNK) {
    __syncthreads();   // previous chunk's LDS reads done
    for (int i = threadIdx.x; i < TCHUNK * 2 * D_STATE; i += 256) {
      int tt = i >> 5, c = i & 31;
      float v = xdbl[(size_t)(b * SEQL + t0 + tt) * 96 + DT_RANK + c];
      if (c < D_STATE) Bs[tt][c] = v; else Cs[tt][c - D_STATE] = v;
    }
    __syncthreads();

    for (int ti = 0; ti < TCHUNK; ++ti) {
      const size_t row = (size_t)(b * SEQL + t0 + ti);
      float dtv = dt_lin[row * D_INNER + d] + bdt;
      dtv = (dtv > 20.0f) ? dtv : log1pf(__expf(dtv));     // softplus
      const float xv  = xconv[row * D_INNER + d];
      const float dtx = dtv * xv;
      float y = 0.0f;
#pragma unroll
      for (int n = 0; n < D_STATE; ++n) {
        float dA = __expf(Arow[n] * dtv);
        h[n] = dA * h[n] + dtx * Bs[ti][n];
        y += h[n] * Cs[ti][n];
      }
      const float yssm = y + xv * Dv;
      const float z    = xz[row * (2 * D_INNER) + D_INNER + d];
      ygate[row * D_INNER + d] = f32_to_bf16(yssm * silu_f(z));
    }
  }
}

// ---------------------------------------------------------------------------
// Host orchestration
// ---------------------------------------------------------------------------
static inline size_t align256(size_t x) { return (x + 255) & ~(size_t)255; }

extern "C" void kernel_launch(void* const* d_in, const int* in_sizes, int n_in,
                              void* d_out, int out_size, void* d_ws, size_t ws_size,
                              hipStream_t stream) {
  (void)in_sizes; (void)n_in; (void)out_size; (void)ws_size;

  const float* x      = (const float*)d_in[0];   // (B,L,1024)
  const float* W_in   = (const float*)d_in[1];   // (1024,4096)
  const float* conv_w = (const float*)d_in[2];   // (2048,1,4)
  const float* conv_b = (const float*)d_in[3];   // (2048)
  const float* W_x    = (const float*)d_in[4];   // (2048,96)
  const float* W_dt   = (const float*)d_in[5];   // (64,2048)
  const float* b_dt   = (const float*)d_in[6];   // (2048)
  const float* W_out  = (const float*)d_in[7];   // (2048,1024)
  const float* A_log  = (const float*)d_in[8];   // (2048,16)
  const float* Dvec   = (const float*)d_in[9];   // (2048)
  float* out = (float*)d_out;                    // (B,L,1024) f32

  char* ws = (char*)d_ws;
  size_t off = 0;
  auto carve = [&](size_t bytes) { size_t o = off; off += align256(bytes); return o; };

  unsigned short* W_in_bf  = (unsigned short*)(ws + carve((size_t)D_MODEL * 2 * D_INNER * 2));
  unsigned short* W_x_bf   = (unsigned short*)(ws + carve((size_t)D_INNER * 96 * 2));
  unsigned short* W_dt_bf  = (unsigned short*)(ws + carve((size_t)DT_RANK * D_INNER * 2));
  unsigned short* W_out_bf = (unsigned short*)(ws + carve((size_t)D_INNER * D_MODEL * 2));
  unsigned short* x_bf     = (unsigned short*)(ws + carve((size_t)NTOK * D_MODEL * 2));
  float*          xz       = (float*)(ws + carve((size_t)NTOK * 2 * D_INNER * 4));
  float*          xconv    = (float*)(ws + carve((size_t)NTOK * D_INNER * 4));
  unsigned short* xconv_bf = (unsigned short*)(ws + carve((size_t)NTOK * D_INNER * 2));
  float*          xdbl     = (float*)(ws + carve((size_t)NTOK * 96 * 4));
  unsigned short* dtA_bf   = (unsigned short*)(ws + carve((size_t)NTOK * DT_RANK * 2));
  float*          dt_lin   = (float*)(ws + carve((size_t)NTOK * D_INNER * 4));
  unsigned short* ygate    = (unsigned short*)(ws + carve((size_t)NTOK * D_INNER * 2));

  auto cvt = [&](const float* src, unsigned short* dst, int n) {
    cvt_f32_bf16<<<(n + 255) / 256, 256, 0, stream>>>(src, dst, n);
  };

  // 1) weight + activation conversions to bf16
  cvt(W_in,  W_in_bf,  D_MODEL * 2 * D_INNER);
  cvt(W_x,   W_x_bf,   D_INNER * 96);
  cvt(W_dt,  W_dt_bf,  DT_RANK * D_INNER);
  cvt(W_out, W_out_bf, D_INNER * D_MODEL);
  cvt(x,     x_bf,     NTOK * D_MODEL);

  // 2) xz = x @ W_in : (4096 x 4096 x 1024)
  gemm_bf16_wmma<<<dim3(NTOK / 128, (2 * D_INNER) / 64), 256, 0, stream>>>(
      x_bf, W_in_bf, xz, NTOK, 2 * D_INNER, D_MODEL);

  // 3) causal depthwise conv + SiLU
  {
    int n = NTOK * D_INNER;
    conv_silu<<<(n + 255) / 256, 256, 0, stream>>>(xz, conv_w, conv_b, xconv, xconv_bf);
  }

  // 4) x_dbl = x_conv @ W_x : (4096 x 96 x 2048)
  gemm_bf16_wmma<<<dim3(NTOK / 128, (96 + 63) / 64), 256, 0, stream>>>(
      xconv_bf, W_x_bf, xdbl, NTOK, 96, D_INNER);

  // 5) dt_lin = dt @ W_dt : (4096 x 2048 x 64)
  {
    int n = NTOK * DT_RANK;
    extract_dt_bf16<<<(n + 255) / 256, 256, 0, stream>>>(xdbl, dtA_bf, NTOK);
  }
  gemm_bf16_wmma<<<dim3(NTOK / 128, D_INNER / 64), 256, 0, stream>>>(
      dtA_bf, W_dt_bf, dt_lin, NTOK, D_INNER, DT_RANK);

  // 6) fused selective scan + gating -> ygate (bf16)
  scan_fused<<<dim3(D_INNER / 256, BATCH), 256, 0, stream>>>(
      xdbl, dt_lin, b_dt, xconv, xz, A_log, Dvec, ygate);

  // 7) out = ygate @ W_out : (4096 x 1024 x 2048)
  gemm_bf16_wmma<<<dim3(NTOK / 128, D_MODEL / 64), 256, 0, stream>>>(
      ygate, W_out_bf, out, NTOK, D_MODEL, D_INNER);
}
